// VQEmbedding_42700564857163
// MI455X (gfx1250) — matrix operand order repaired and checked by reading, not compiled, not run on hardware
//
#include <hip/hip_runtime.h>
#include <hip/hip_bf16.h>

typedef __attribute__((ext_vector_type(16))) _Float16 v16h;
typedef __attribute__((ext_vector_type(8)))  float    v8f;

#define KCB 8192   // codebook entries
#define DIM 512    // embedding dim
#define BN  32768  // B*N rows

#define FLT_BIG 3.402823466e+38f

// ---------------------------------------------------------------------------
// Kernel 1: squared norms of codebook rows. One wave32 per row, float4 loads,
// butterfly reduce.
// ---------------------------------------------------------------------------
__global__ void __launch_bounds__(256) vq_norms(const float* __restrict__ cb,
                                                float* __restrict__ norms) {
  const int wave = (blockIdx.x * blockDim.x + threadIdx.x) >> 5;  // row
  const int lane = threadIdx.x & 31;
  if (wave >= KCB) return;
  const float4* row = (const float4*)(cb + (size_t)wave * DIM);
  float s = 0.f;
#pragma unroll
  for (int i = 0; i < 4; ++i) {
    float4 v = row[lane + i * 32];
    s += v.x * v.x + v.y * v.y + v.z * v.z + v.w * v.w;
  }
#pragma unroll
  for (int off = 16; off > 0; off >>= 1) s += __shfl_xor(s, off, 32);
  if (lane == 0) norms[wave] = s;
}

// ---------------------------------------------------------------------------
// Kernel 2: repack codebook (fp32, row-major [K][D]) into fp16 in WMMA
// B-operand order: element index = ((kb*16 + c)*32 + l)*16 + h
//   kb = k-slice (16 columns), c = 32-deep d-chunk, l = lane (l%16 = output column
//   within slice), h = element (with l/16 gives d within chunk).
// Hot-loop B load becomes one contiguous v16h (32B) per lane, coalesced.
// ---------------------------------------------------------------------------
__global__ void __launch_bounds__(256) vq_pack(const float* __restrict__ cb,
                                               _Float16* __restrict__ packB) {
  const int gid = blockIdx.x * blockDim.x + threadIdx.x;  // 0 .. K*D-1
  const int h  = gid & 15;
  const int l  = (gid >> 4) & 31;
  const int c  = (gid >> 9) & 15;
  const int kb = gid >> 13;
  packB[gid] = (_Float16)cb[(size_t)(kb * 16 + (l & 15)) * DIM + c * 32 + ((h & 8) << 1) + ((l >> 4) << 3) + (h & 7)];
}

// ---------------------------------------------------------------------------
// Kernel 3: fused distance GEMM + argmin + gather.
// One workgroup (8 waves) handles 16 x-rows against all 8192 codes.
// ---------------------------------------------------------------------------
__global__ void __launch_bounds__(256) vq_main(
    const float* __restrict__ z, const float* __restrict__ cb,
    const float* __restrict__ norms, const _Float16* __restrict__ packB,
    float* __restrict__ out_codes, float* __restrict__ out_tilde,
    float* __restrict__ out_idx) {
  __shared__ v16h  ldsA[16][32];   // 16 d-chunks x 32 lanes, fp16 A layout (16KB)
  __shared__ float redV[8][16];    // per-wave per-row best value
  __shared__ int   redI[8][16];    // per-wave per-row best index
  __shared__ int   fIdx[16];       // final index per row

  const int tid  = threadIdx.x;
  const int lane = tid & 31;
  const int wave = tid >> 5;
  const int row0 = blockIdx.x * 16;

  // ---- Stage A tile (16 rows x 512) into LDS, fp16, pre-swizzled into the
  // ISA A layout: lane m=l%16 holds row m; halves 0..7 = d {hi*8+0..7},
  // halves 8..15 = d {16+hi*8+0..7} within each 32-deep chunk (hi = l/16).
#pragma unroll
  for (int s = tid; s < 512; s += 256) {
    const int c = s >> 5, l = s & 31, m = l & 15, hi = l >> 4;
    const float* src = z + (size_t)(row0 + m) * DIM + c * 32 + hi * 8;
    float4 f0 = *(const float4*)(src);
    float4 f1 = *(const float4*)(src + 4);
    float4 f2 = *(const float4*)(src + 16);
    float4 f3 = *(const float4*)(src + 20);
    v16h a;
    a[0]  = (_Float16)f0.x; a[1]  = (_Float16)f0.y;
    a[2]  = (_Float16)f0.z; a[3]  = (_Float16)f0.w;
    a[4]  = (_Float16)f1.x; a[5]  = (_Float16)f1.y;
    a[6]  = (_Float16)f1.z; a[7]  = (_Float16)f1.w;
    a[8]  = (_Float16)f2.x; a[9]  = (_Float16)f2.y;
    a[10] = (_Float16)f2.z; a[11] = (_Float16)f2.w;
    a[12] = (_Float16)f3.x; a[13] = (_Float16)f3.y;
    a[14] = (_Float16)f3.z; a[15] = (_Float16)f3.w;
    ldsA[c][l] = a;
  }
  __syncthreads();

  // ---- Running per-row argmin. C layout: lane L, vgpr v -> M = v + 8*(L/16),
  // N = L%16; so each lane tracks 8 rows for one column slot per iteration.
  float minv[8];
  int   mini[8];
#pragma unroll
  for (int v = 0; v < 8; ++v) { minv[v] = FLT_BIG; mini[v] = 0; }

  const int ncol = lane & 15;
  for (int kk = 0; kk < KCB / 128; ++kk) {
    const int k0 = kk * 128 + wave * 16;  // this wave's 16-column slice
    v8f acc = {};
    const v16h* bp = (const v16h*)packB + (size_t)(k0 >> 4) * 512 + lane;
    __builtin_prefetch((const void*)(bp + 8 * 512), 0, 0);  // next K slice
#pragma unroll
    for (int c = 0; c < 16; ++c) {
      v16h b = bp[c * 32];        // coalesced 32B/lane global load (L2-resident)
      v16h a = ldsA[c][lane];     // ds_load_b128 x2
      acc = __builtin_amdgcn_wmma_f32_16x16x32_f16(
          /*neg_a=*/false, a, /*neg_b=*/false, b,
          /*c_mod=*/(short)0, acc, /*reuse_a=*/false, /*reuse_b=*/false);
    }
    const int   kcol = k0 + ncol;
    const float nv   = norms[kcol];
#pragma unroll
    for (int v = 0; v < 8; ++v) {
      float d = nv + 2.0f * acc[v];  // ||c||^2 + 2 x.c  (||x||^2 const/row)
      if (d < minv[v]) { minv[v] = d; mini[v] = kcol; }
    }
  }

  // ---- Reduce across the 16 lanes sharing each row (tie -> lowest index).
#pragma unroll
  for (int v = 0; v < 8; ++v) {
    float val = minv[v];
    int   idx = mini[v];
#pragma unroll
    for (int off = 8; off > 0; off >>= 1) {
      float ov = __shfl_xor(val, off, 16);
      int   oi = __shfl_xor(idx, off, 16);
      if (ov < val || (ov == val && oi < idx)) { val = ov; idx = oi; }
    }
    if (ncol == 0) {
      const int r = v + (lane >> 4) * 8;
      redV[wave][r] = val;
      redI[wave][r] = idx;
    }
  }
  __syncthreads();

  // ---- Combine the 8 waves' candidates; emit index.
  if (tid < 16) {
    float bv = redV[0][tid];
    int   bi = redI[0][tid];
#pragma unroll
    for (int w = 1; w < 8; ++w) {
      float ov = redV[w][tid];
      int   oi = redI[w][tid];
      if (ov < bv || (ov == bv && oi < bi)) { bv = ov; bi = oi; }
    }
    fIdx[tid] = bi;
    out_idx[row0 + tid] = (float)bi;
  }
  __syncthreads();

  // ---- Gather winning codebook rows (fp32) to both outputs.
  {
    const int r = tid >> 4, p = tid & 15;
    const int bi = fIdx[r];
    const float4* src = (const float4*)(cb + (size_t)bi * DIM);
    float4* d0 = (float4*)(out_codes + (size_t)(row0 + r) * DIM);
    float4* d1 = (float4*)(out_tilde + (size_t)(row0 + r) * DIM);
#pragma unroll
    for (int i = 0; i < 8; ++i) {
      float4 v = src[p + i * 16];
      d0[p + i * 16] = v;
      d1[p + i * 16] = v;
    }
  }
}

// ---------------------------------------------------------------------------
extern "C" void kernel_launch(void* const* d_in, const int* in_sizes, int n_in,
                              void* d_out, int out_size, void* d_ws,
                              size_t ws_size, hipStream_t stream) {
  const float* z  = (const float*)d_in[0];  // [8,4096,512] fp32
  const float* cb = (const float*)d_in[1];  // [8192,512]  fp32

  float*     norms = (float*)d_ws;                          // 32 KB
  _Float16*  packB = (_Float16*)((char*)d_ws + 32 * 1024);  // 8 MB

  float* out_codes = (float*)d_out;
  float* out_tilde = out_codes + (size_t)BN * DIM;
  float* out_idx   = out_tilde + (size_t)BN * DIM;

  vq_norms<<<KCB / 8, 256, 0, stream>>>(cb, norms);
  vq_pack<<<(KCB * DIM) / 256, 256, 0, stream>>>(cb, packB);
  vq_main<<<BN / 16, 256, 0, stream>>>(z, cb, norms, packB, out_codes,
                                       out_tilde, out_idx);
}
